// TrittentionCScore_2121713844818
// MI455X (gfx1250) — matrix operand order repaired and checked by reading, not compiled, not run on hardware
//
#include <hip/hip_runtime.h>

// B=1, S=256, H=8, D=64
//   c,b,a : [256][8][64] f32      W : [8][64][64][64] f32
//   out   : [8][256][256][256] f32  (n,p,q,r)
//
// Pipeline (f16 WMMA, f32 accum):
//   K1: bsum[q][j]   = sum_n b[q,n,j]                       (f16, 32KB ws)
//   K2: s1[r][i*64+j]= sum_{n,k} c[r,nk] * W[n,i,j,k]       (f16, 2MB ws, WMMA)
//   K3: s2t[(q*256+r)][i] = sum_j bsum[q,j]*s1[r][i][j]     (f16, 8MB ws, WMMA)
//   K4: out[n][p][q*256+r] = sum_i a[p,n,i]*s2t[qr][i]      (f32, WMMA, NT stores)

typedef _Float16 v16h __attribute__((ext_vector_type(16)));
typedef _Float16 h2   __attribute__((ext_vector_type(2)));
typedef float    v8f  __attribute__((ext_vector_type(8)));

__global__ __launch_bounds__(256) void k_bsum(const float* __restrict__ b,
                                              _Float16* __restrict__ bsum) {
  int idx = blockIdx.x * 256 + threadIdx.x;   // q*64 + j, 16384 total
  int q = idx >> 6, j = idx & 63;
  float s = 0.f;
#pragma unroll
  for (int n = 0; n < 8; ++n) s += b[q * 512 + n * 64 + j];
  bsum[idx] = (_Float16)s;
}

// step1: M=r(256), N=ij(4096), K=nk(512). Wave = one r-tile x 4 ij-tiles.
__global__ __launch_bounds__(256) void k_step1(const float* __restrict__ c,
                                               const float* __restrict__ W,
                                               _Float16* __restrict__ s1) {
  const int lane  = threadIdx.x & 31;
  const int laneN = lane & 15;
  const int half  = lane >> 4;
  const int w   = blockIdx.x * 8 + (threadIdx.x >> 5);  // 0..1023
  const int rt  = w >> 6;                               // 0..15
  const int ijg = w & 63;                               // 0..63 (4 tiles each)
  const int r   = rt * 16 + laneN;

  v8f acc[4] = {v8f{}, v8f{}, v8f{}, v8f{}};

#pragma unroll 1
  for (int ks = 0; ks < 16; ++ks) {
    const int nk0 = ks * 32;
    // A fragment: c row r, contiguous K=e+16*half, same order as B
    v16h afr;
#pragma unroll
    for (int p = 0; p < 8; ++p) {
      const int kk = 2 * p + 16 * half + nk0;
      float2 cv = *(const float2*)(c + r * 512 + kk);
      afr[2 * p]     = (_Float16)cv.x;
      afr[2 * p + 1] = (_Float16)cv.y;
    }
    // B fragment: W[nk, ij], flat = n*262144 + ij*64 + k ; 16 contiguous k per lane
    const int nkL = nk0 + 16 * half;
    const int n   = nkL >> 6;
    const int k0  = nkL & 63;
    const float* wbase = W + n * 262144 + k0;
#pragma unroll
    for (int t = 0; t < 4; ++t) {
      const int ij = ijg * 64 + t * 16 + laneN;
      const float4* wp = (const float4*)(wbase + ij * 64);
      v16h bfr;
#pragma unroll
      for (int u = 0; u < 4; ++u) {
        float4 f = wp[u];
        bfr[4 * u + 0] = (_Float16)f.x;
        bfr[4 * u + 1] = (_Float16)f.y;
        bfr[4 * u + 2] = (_Float16)f.z;
        bfr[4 * u + 3] = (_Float16)f.w;
      }
      acc[t] = __builtin_amdgcn_wmma_f32_16x16x32_f16(false, afr, false, bfr,
                                                      (short)0, acc[t], false, false);
    }
  }
  // D layout: M = v + 8*half, N = laneN. Store f16 as s1[r][ij].
#pragma unroll
  for (int t = 0; t < 4; ++t) {
    const int ij = ijg * 64 + t * 16 + laneN;
#pragma unroll
    for (int v = 0; v < 8; ++v) {
      const int rr = rt * 16 + v + 8 * half;
      s1[rr * 4096 + ij] = (_Float16)acc[t][v];
    }
  }
}

// step2: per r: D[q,i] = bsum(q x j) * s1_r^T(j x i). Wave = (r, q-tile), 4 i-tiles.
__global__ __launch_bounds__(256) void k_step2(const _Float16* __restrict__ bsum,
                                               const _Float16* __restrict__ s1,
                                               _Float16* __restrict__ s2t) {
  const int lane  = threadIdx.x & 31;
  const int laneN = lane & 15;
  const int half  = lane >> 4;
  const int w  = blockIdx.x * 8 + (threadIdx.x >> 5);  // 0..4095
  const int r  = w >> 4;
  const int qt = w & 15;

  v16h afr[2];
#pragma unroll
  for (int ks = 0; ks < 2; ++ks) {
#pragma unroll
    for (int p = 0; p < 8; ++p) {
      const int kk = 2 * p + 16 * half + 32 * ks;
      h2 t2 = *(const h2*)(bsum + (qt * 16 + laneN) * 64 + kk);
      afr[ks][2 * p]     = t2.x;
      afr[ks][2 * p + 1] = t2.y;
    }
  }
#pragma unroll
  for (int t = 0; t < 4; ++t) {
    const int i0 = t * 16;
    v8f acc = {};
#pragma unroll
    for (int ks = 0; ks < 2; ++ks) {
      // B[j,i]: N=i=i0+laneN, K=j=ks*32+16*half+e -> 32B contiguous in s1
      v16h bfr = *(const v16h*)(s1 + r * 4096 + (i0 + laneN) * 64 + ks * 32 + 16 * half);
      acc = __builtin_amdgcn_wmma_f32_16x16x32_f16(false, afr[ks], false, bfr,
                                                   (short)0, acc, false, false);
    }
#pragma unroll
    for (int v = 0; v < 8; ++v) {
      const int q = qt * 16 + v + 8 * half;
      s2t[(q * 256 + r) * 64 + i0 + laneN] = (_Float16)acc[v];
    }
  }
}

// step3: per head n: out_n[p, qr] = a_n(p x i) * s2t^T(i x qr). A held in regs, 32 qr-tiles/wave.
__global__ __launch_bounds__(256) void k_step3(const float* __restrict__ a,
                                               const _Float16* __restrict__ s2t,
                                               float* __restrict__ out) {
  const int lane  = threadIdx.x & 31;
  const int laneN = lane & 15;
  const int half  = lane >> 4;
  const int w     = blockIdx.x * 8 + (threadIdx.x >> 5);  // 0..16383
  const int g     = w & 127;        // (n, p-tile)
  const int chunk = w >> 7;         // 0..127 -> 32 qr-tiles each
  const int n  = g >> 4;
  const int pt = g & 15;

  v16h afr[2];
#pragma unroll
  for (int ks = 0; ks < 2; ++ks) {
#pragma unroll
    for (int p = 0; p < 8; ++p) {
      const int kk = 2 * p + 16 * half + 32 * ks;
      float2 av = *(const float2*)(a + ((pt * 16 + laneN) * 8 + n) * 64 + kk);
      afr[ks][2 * p]     = (_Float16)av.x;
      afr[ks][2 * p + 1] = (_Float16)av.y;
    }
  }
#pragma unroll 1
  for (int qi = 0; qi < 32; ++qi) {
    const int qr0 = (chunk * 32 + qi) * 16;
    v8f acc = {};
#pragma unroll
    for (int ks = 0; ks < 2; ++ks) {
      // B[i,qr]: N=qr0+laneN, K=i=ks*32+16*half+e -> one aligned 32B v16h load
      v16h bfr = *(const v16h*)(s2t + (qr0 + laneN) * 64 + ks * 32 + 16 * half);
      acc = __builtin_amdgcn_wmma_f32_16x16x32_f16(false, afr[ks], false, bfr,
                                                   (short)0, acc, false, false);
    }
    float* o = out + ((size_t)n << 24) + (size_t)(qr0 + laneN);
#pragma unroll
    for (int v = 0; v < 8; ++v) {
      __builtin_nontemporal_store(acc[v], o + ((size_t)(pt * 16 + v + 8 * half) << 16));
    }
  }
}

extern "C" void kernel_launch(void* const* d_in, const int* in_sizes, int n_in,
                              void* d_out, int out_size, void* d_ws, size_t ws_size,
                              hipStream_t stream) {
  const float* c = (const float*)d_in[0];
  const float* b = (const float*)d_in[1];
  const float* a = (const float*)d_in[2];
  const float* W = (const float*)d_in[3];
  float* out = (float*)d_out;

  _Float16* bsum = (_Float16*)d_ws;        // 16384 halves (32 KB)
  _Float16* s1   = bsum + 16384;           // 1,048,576 halves (2 MB)
  _Float16* s2t  = s1 + 1048576;           // 4,194,304 halves (8 MB)

  k_bsum <<<  64, 256, 0, stream>>>(b, bsum);
  k_step1<<< 128, 256, 0, stream>>>(c, W, s1);
  k_step2<<< 512, 256, 0, stream>>>(bsum, s1, s2t);
  k_step3<<<2048, 256, 0, stream>>>(a, s2t, out);
}
